// EnetGnn_24859270710016
// MI455X (gfx1250) — compile-verified, matrix operand order checked by reading, not run on hardware
//
#include <hip/hip_runtime.h>
#include <hip/hip_bf16.h>

// ---------------- problem constants (from setup_inputs) ----------------
#define NB   8      // batch
#define CH   128    // feature channels
#define HH   45
#define WW   60
#define HWP  (HH*WW)        // 2700 nodes per image
#define KNN_K 16
#define HI   (HH*8)         // 360
#define WI   (WW*8)         // 480
#define GNN_ITERS 2

typedef __attribute__((ext_vector_type(16))) __bf16 v16bf;
typedef __attribute__((ext_vector_type(8)))  __bf16 v8bf;
typedef __attribute__((ext_vector_type(8)))  float  v8f;

static __device__ __forceinline__ v16bf bf_zero16() {
    v16bf z;
#pragma unroll
    for (int i = 0; i < 16; ++i) z[i] = (__bf16)0.0f;
    return z;
}

// Load one lane's A/B fragment for v_wmma_f32_16x16x32_bf16.
// Per ISA layout: lanes 0-15 (kh=0) hold K = [ksBase..+7] and [ksBase+16..+23],
// lanes 16-31 (kh=1) hold K = [ksBase+8..+15] and [ksBase+24..+31].
static __device__ __forceinline__ v16bf load_frag(const __bf16* rowBase, int ksBase, int kh) {
    v8bf lo = *(const v8bf*)(rowBase + ksBase + kh * 8);
    v8bf hi = *(const v8bf*)(rowBase + ksBase + 16 + kh * 8);
    v16bf r;
#pragma unroll
    for (int i = 0; i < 8; ++i) { r[i] = lo[i]; r[8 + i] = hi[i]; }
    return r;
}

// ---------------- pack kernels ----------------
__global__ void pack_mat_kernel(const float* __restrict__ src, __bf16* __restrict__ dst, int count) {
    int i = blockIdx.x * blockDim.x + threadIdx.x;
    if (i < count) dst[i] = (__bf16)src[i];
}

// conv_w (O=128, I=256, 3, 3) -> cwpk[tap][o][i]  (tap = dy*3+dx)
__global__ void pack_conv_kernel(const float* __restrict__ w, __bf16* __restrict__ dst) {
    int e = blockIdx.x * blockDim.x + threadIdx.x;
    const int total = 9 * CH * 2 * CH;
    if (e >= total) return;
    int tap = e / (CH * 2 * CH);
    int rem = e % (CH * 2 * CH);
    int o = rem / (2 * CH);
    int i = rem % (2 * CH);
    dst[e] = (__bf16)w[((size_t)o * (2 * CH) + i) * 9 + tap];
}

// cnn (N,C,H,W) -> h0 (N, HW, C)
__global__ void transpose_h_kernel(const float* __restrict__ cnn, float* __restrict__ h0) {
    size_t e = (size_t)blockIdx.x * blockDim.x + threadIdx.x;
    const size_t total = (size_t)NB * HWP * CH;
    if (e >= total) return;
    int c = (int)(e % CH);
    size_t node = e / CH;
    int p = (int)(node % HWP);
    int n = (int)(node / HWP);
    h0[e] = cnn[((size_t)n * CH + c) * HWP + p];
}

// ---------------- median pool (lower median of 8x8 block) ----------------
__global__ void median_kernel(const float* __restrict__ orig, const float* __restrict__ xy,
                              float* __restrict__ proj) {
    __shared__ float sv[64][65];
    int gid = blockIdx.x * 64 + threadIdx.x;
    const int total = NB * HWP * 3;
    if (gid >= total) return;
    int comp = gid % 3;
    int p = (gid / 3) % HWP;
    int n = gid / (3 * HWP);
    int y = p / WW, x = p % WW;
    const float* src;
    if (comp == 0)      src = xy   + ((size_t)n * 2 + 0) * HI * WI;
    else if (comp == 1) src = xy   + ((size_t)n * 2 + 1) * HI * WI;
    else                src = orig + ((size_t)n * 4 + 3) * HI * WI;
    float* v = sv[threadIdx.x];
#pragma unroll
    for (int dy = 0; dy < 8; ++dy)
#pragma unroll
        for (int dx = 0; dx < 8; ++dx)
            v[dy * 8 + dx] = src[(size_t)(y * 8 + dy) * WI + (x * 8 + dx)];
    // lower median: element whose sorted position range covers index 31
    float med = v[0];
    for (int i = 0; i < 64; ++i) {
        float vi = v[i];
        int cl = 0, ce = 0;
        for (int j = 0; j < 64; ++j) { cl += (v[j] < vi); ce += (v[j] == vi); }
        if (cl <= 31 && 31 < cl + ce) { med = vi; break; }
    }
    proj[gid] = med;
}

// ---------------- KNN: one wave (32 lanes) per query row ----------------
__global__ void knn_kernel(const float* __restrict__ proj, int* __restrict__ knn) {
    int gtid = blockIdx.x * blockDim.x + threadIdx.x;
    int wave = gtid >> 5;
    int lane = threadIdx.x & 31;
    if (wave >= NB * HWP) return;
    int n = wave / HWP;
    int i = wave % HWP;
    const float* pb = proj + (size_t)n * HWP * 3;
    float xi = pb[(size_t)i * 3 + 0];
    float yi = pb[(size_t)i * 3 + 1];
    float zi = pb[(size_t)i * 3 + 2];

    const float INF = 3.0e38f;
    float d[KNN_K]; int id[KNN_K];
#pragma unroll
    for (int s = 0; s < KNN_K; ++s) { d[s] = INF; id[s] = 0x7fffffff; }

    for (int j = lane; j < HWP; j += 32) {
        float dx = pb[(size_t)j * 3 + 0] - xi;
        float dy = pb[(size_t)j * 3 + 1] - yi;
        float dz = pb[(size_t)j * 3 + 2] - zi;
        float nd = dx * dx + dy * dy + dz * dz;
        int ni = j;
        // sorted insert (fully unrolled, register-resident)
#pragma unroll
        for (int s = 0; s < KNN_K; ++s) {
            bool less = (nd < d[s]) || (nd == d[s] && ni < id[s]);
            float td = d[s]; int ti = id[s];
            d[s]  = less ? nd : td;  id[s] = less ? ni : ti;
            nd    = less ? td : nd;  ni    = less ? ti : ni;
        }
    }
    // merge across the wave: 16 rounds of wave-min over the lane heads
    for (int r = 0; r < KNN_K; ++r) {
        float bd = d[0]; int bi = id[0];
#pragma unroll
        for (int off = 16; off > 0; off >>= 1) {
            float od = __shfl_xor(bd, off, 32);
            int   oi = __shfl_xor(bi, off, 32);
            if (od < bd || (od == bd && oi < bi)) { bd = od; bi = oi; }
        }
        bool win = (d[0] == bd) && (id[0] == bi);
#pragma unroll
        for (int s = 0; s < KNN_K - 1; ++s) {
            d[s]  = win ? d[s + 1]  : d[s];
            id[s] = win ? id[s + 1] : id[s];
        }
        if (win) { d[KNN_K - 1] = INF; id[KNN_K - 1] = 0x7fffffff; }
        if (lane == 0) knn[(size_t)wave * KNN_K + r] = bi;
    }
}

// ---------------- GNN: gather + two Linear(128,128)+PReLU + mean ----------------
// 1 block = 1 node; 8 waves each own a 16-wide output-feature tile.
__global__ void gnn_mlp_kernel(const float* __restrict__ h, const int* __restrict__ knn,
                               const __bf16* __restrict__ gw0, const float* __restrict__ gb0,
                               const float* __restrict__ ga0,
                               const __bf16* __restrict__ gw1, const float* __restrict__ gb1,
                               const float* __restrict__ ga1,
                               float* __restrict__ m) {
    __shared__ __align__(16) __bf16 sA[16][CH];
    __shared__ __align__(16) __bf16 sB[16][CH];
    __shared__ float sO[16][CH];

    int node = blockIdx.x;
    int n = node / HWP;
    const float* hb = h + (size_t)n * HWP * CH;
    const int* kn = knn + (size_t)node * KNN_K;
    int tid = threadIdx.x;

    // gather 16 neighbor rows -> bf16 in LDS
    for (int e = tid; e < 16 * CH; e += 256) {
        int r = e >> 7, c = e & 127;
        sA[r][c] = (__bf16)hb[(size_t)kn[r] * CH + c];
    }
    __syncthreads();

    int wave = tid >> 5, lane = tid & 31;
    int mr = lane & 15, kh = lane >> 4;
    int nc = wave * 16 + (lane & 15);
    float a0 = ga0[0], a1 = ga1[0];

    // ---- layer 0 ----
    v8f acc = {};
#pragma unroll
    for (int ks = 0; ks < 4; ++ks) {
        v16bf a = load_frag(&sA[mr][0], ks * 32, kh);
        v16bf b = load_frag(gw0 + (size_t)nc * CH, ks * 32, kh);
        acc = __builtin_amdgcn_wmma_f32_16x16x32_bf16(false, a, false, b, (short)0, acc, false, false);
    }
    {
        float bias = gb0[nc];
#pragma unroll
        for (int r = 0; r < 8; ++r) {
            int mm = r + 8 * kh;
            float v = acc[r] + bias;
            v = (v >= 0.0f) ? v : a0 * v;
            sB[mm][nc] = (__bf16)v;
        }
    }
    __syncthreads();

    // ---- layer 1 ----
    v8f acc1 = {};
#pragma unroll
    for (int ks = 0; ks < 4; ++ks) {
        v16bf a = load_frag(&sB[mr][0], ks * 32, kh);
        v16bf b = load_frag(gw1 + (size_t)nc * CH, ks * 32, kh);
        acc1 = __builtin_amdgcn_wmma_f32_16x16x32_bf16(false, a, false, b, (short)0, acc1, false, false);
    }
    {
        float bias = gb1[nc];
#pragma unroll
        for (int r = 0; r < 8; ++r) {
            int mm = r + 8 * kh;
            float v = acc1[r] + bias;
            v = (v >= 0.0f) ? v : a1 * v;
            sO[mm][nc] = v;
        }
    }
    __syncthreads();

    // mean over the 16 neighbors
    if (tid < CH) {
        float s = 0.0f;
#pragma unroll
        for (int r = 0; r < 16; ++r) s += sO[r][tid];
        m[(size_t)node * CH + tid] = s * (1.0f / 16.0f);
    }
}

// ---------------- q-update: prelu(concat(h,m) @ q_w.T + q_b) ----------------
// 1 block = 16 nodes; K = 256.
__global__ void q_update_kernel(const float* __restrict__ h, const float* __restrict__ m,
                                const __bf16* __restrict__ qw, const float* __restrict__ qb,
                                const float* __restrict__ qa, float* __restrict__ hout) {
    __shared__ __align__(16) __bf16 sA[16][2 * CH];
    size_t base = (size_t)blockIdx.x * 16;
    int tid = threadIdx.x;
    for (int e = tid; e < 16 * 2 * CH; e += 256) {
        int r = e >> 8, c = e & 255;
        size_t node = base + r;
        float v = (c < CH) ? h[node * CH + c] : m[node * CH + (c - CH)];
        sA[r][c] = (__bf16)v;
    }
    __syncthreads();

    int wave = tid >> 5, lane = tid & 31;
    int mr = lane & 15, kh = lane >> 4;
    int nc = wave * 16 + (lane & 15);
    float aq = qa[0];

    v8f acc = {};
#pragma unroll
    for (int ks = 0; ks < 8; ++ks) {
        v16bf a = load_frag(&sA[mr][0], ks * 32, kh);
        v16bf b = load_frag(qw + (size_t)nc * (2 * CH), ks * 32, kh);
        acc = __builtin_amdgcn_wmma_f32_16x16x32_bf16(false, a, false, b, (short)0, acc, false, false);
    }
    float bias = qb[nc];
#pragma unroll
    for (int r = 0; r < 8; ++r) {
        int mm = r + 8 * kh;
        float v = acc[r] + bias;
        v = (v >= 0.0f) ? v : aq * v;
        hout[(base + mm) * CH + nc] = v;
    }
}

// ---------------- build channels-last bf16 conv input (N, HW, 2C) ----------------
__global__ void build_inpT_kernel(const float* __restrict__ cnn, const float* __restrict__ h,
                                  __bf16* __restrict__ inpT) {
    size_t e = (size_t)blockIdx.x * blockDim.x + threadIdx.x;
    const size_t total = (size_t)NB * HWP * 2 * CH;
    if (e >= total) return;
    int c = (int)(e % (2 * CH));
    size_t node = e / (2 * CH);
    int p = (int)(node % HWP);
    int n = (int)(node / HWP);
    float v;
    if (c < CH) v = cnn[((size_t)n * CH + c) * HWP + p];
    else        v = h[node * CH + (c - CH)];
    inpT[e] = (__bf16)v;
}

// ---------------- 3x3 conv (2C -> C) as WMMA GEMM, K = 9*256 ----------------
// 1 block = 16 flat pixels of one image; 8 waves cover 128 output channels.
__global__ void conv_kernel(const __bf16* __restrict__ inpT, const __bf16* __restrict__ cwpk,
                            const float* __restrict__ cb, float* __restrict__ out) {
    const int tilesPerImg = (HWP + 15) / 16;   // 169
    int n  = blockIdx.x / tilesPerImg;
    int pt = blockIdx.x % tilesPerImg;
    int tid = threadIdx.x;
    int wave = tid >> 5, lane = tid & 31;
    int mr = lane & 15, kh = lane >> 4;
    int nc = wave * 16 + (lane & 15);

    int p = pt * 16 + mr;
    int y = (p < HWP) ? (p / WW) : 0;
    int x = (p < HWP) ? (p % WW) : 0;
    const __bf16* img = inpT + (size_t)n * HWP * (2 * CH);

    v8f acc = {};
#pragma unroll
    for (int dy = -1; dy <= 1; ++dy) {
#pragma unroll
        for (int dx = -1; dx <= 1; ++dx) {
            int yy = y + dy, xx = x + dx;
            bool ok = (p < HWP) && (yy >= 0) && (yy < HH) && (xx >= 0) && (xx < WW);
            int srcp = ok ? (yy * WW + xx) : 0;
            const __bf16* row  = img + (size_t)srcp * (2 * CH);
            int tap = (dy + 1) * 3 + (dx + 1);
            const __bf16* wrow = cwpk + ((size_t)tap * CH + nc) * (2 * CH);
#pragma unroll
            for (int ks = 0; ks < 8; ++ks) {
                v16bf a = load_frag(row, ks * 32, kh);
                if (!ok) a = bf_zero16();
                v16bf b = load_frag(wrow, ks * 32, kh);
                acc = __builtin_amdgcn_wmma_f32_16x16x32_bf16(false, a, false, b, (short)0, acc, false, false);
            }
        }
    }
    float bias = cb[nc];
#pragma unroll
    for (int r = 0; r < 8; ++r) {
        int mm = r + 8 * kh;
        int pp = pt * 16 + mm;
        if (pp < HWP)
            out[((size_t)n * CH + nc) * HWP + pp] = acc[r] + bias;
    }
}

// ---------------- host launch ----------------
extern "C" void kernel_launch(void* const* d_in, const int* in_sizes, int n_in,
                              void* d_out, int out_size, void* d_ws, size_t ws_size,
                              hipStream_t stream) {
    const float* cnn    = (const float*)d_in[0];
    const float* orig   = (const float*)d_in[1];
    const float* xy     = (const float*)d_in[2];
    const float* g_w0   = (const float*)d_in[3];
    const float* g_b0   = (const float*)d_in[4];
    const float* g_a0   = (const float*)d_in[5];
    const float* g_w1   = (const float*)d_in[6];
    const float* g_b1   = (const float*)d_in[7];
    const float* g_a1   = (const float*)d_in[8];
    const float* q_w    = (const float*)d_in[9];
    const float* q_b    = (const float*)d_in[10];
    const float* q_a    = (const float*)d_in[11];
    const float* conv_w = (const float*)d_in[12];
    const float* conv_b = (const float*)d_in[13];
    float* out = (float*)d_out;

    char* ws = (char*)d_ws;
    size_t off = 0;
    auto alloc = [&](size_t bytes) -> char* {
        char* p = ws + off;
        off = (off + bytes + 255) & ~(size_t)255;
        return p;
    };
    float*  proj = (float*)alloc((size_t)NB * HWP * 3 * 4);
    int*    knn  = (int*)  alloc((size_t)NB * HWP * KNN_K * 4);
    float*  hA   = (float*)alloc((size_t)NB * HWP * CH * 4);
    float*  hB   = (float*)alloc((size_t)NB * HWP * CH * 4);
    float*  mbuf = (float*)alloc((size_t)NB * HWP * CH * 4);
    __bf16* inpT = (__bf16*)mbuf;   // alias: m dead before inpT is built (same byte size)
    __bf16* gw0  = (__bf16*)alloc((size_t)CH * CH * 2);
    __bf16* gw1  = (__bf16*)alloc((size_t)CH * CH * 2);
    __bf16* qw   = (__bf16*)alloc((size_t)CH * 2 * CH * 2);
    __bf16* cwpk = (__bf16*)alloc((size_t)9 * CH * 2 * CH * 2);

    // pack weights to bf16
    pack_mat_kernel<<<(CH * CH + 255) / 256, 256, 0, stream>>>(g_w0, gw0, CH * CH);
    pack_mat_kernel<<<(CH * CH + 255) / 256, 256, 0, stream>>>(g_w1, gw1, CH * CH);
    pack_mat_kernel<<<(CH * 2 * CH + 255) / 256, 256, 0, stream>>>(q_w, qw, CH * 2 * CH);
    pack_conv_kernel<<<(9 * CH * 2 * CH + 255) / 256, 256, 0, stream>>>(conv_w, cwpk);

    // h0 = cnn transposed to (N, HW, C)
    {
        size_t total = (size_t)NB * HWP * CH;
        transpose_h_kernel<<<(int)((total + 255) / 256), 256, 0, stream>>>(cnn, hA);
    }

    // median pool -> proj (N, HW, 3)
    median_kernel<<<(NB * HWP * 3 + 63) / 64, 64, 0, stream>>>(orig, xy, proj);

    // knn (k = 16)
    knn_kernel<<<(NB * HWP + 7) / 8, 256, 0, stream>>>(proj, knn);

    // GNN iterations (ping-pong hA <-> hB)
    float* hcur = hA;
    float* hnxt = hB;
    for (int it = 0; it < GNN_ITERS; ++it) {
        gnn_mlp_kernel<<<NB * HWP, 256, 0, stream>>>(hcur, knn, gw0, g_b0, g_a0,
                                                     gw1, g_b1, g_a1, mbuf);
        q_update_kernel<<<NB * HWP / 16, 256, 0, stream>>>(hcur, mbuf, qw, q_b, q_a, hnxt);
        float* t = hcur; hcur = hnxt; hnxt = t;
    }

    // channels-last bf16 conv input (overwrites mbuf alias)
    {
        size_t total = (size_t)NB * HWP * 2 * CH;
        build_inpT_kernel<<<(int)((total + 255) / 256), 256, 0, stream>>>(cnn, hcur, inpT);
    }

    // 3x3 conv -> d_out (N, C, H, W)
    conv_kernel<<<NB * ((HWP + 15) / 16), 256, 0, stream>>>(inpT, cwpk, conv_b, out);
}
